// StereoGNNOrdinal_15710990368932
// MI455X (gfx1250) — matrix operand order repaired and checked by reading, not compiled, not run on hardware
//
#include <hip/hip_runtime.h>
#include <hip/hip_bf16.h>
#include <math.h>

// ---------------- problem constants ----------------
#define NNODES 50000
#define NEDGES 800000
#define NGRAPH 2048
#define NODE_DIM 86
#define EDGE_DIM 18
#define HID 128
#define EHID 64
#define SLOPE 0.2f

typedef __attribute__((ext_vector_type(16))) _Float16 v16h;
typedef __attribute__((ext_vector_type(8)))  _Float16 h8;
typedef __attribute__((ext_vector_type(8)))  float    v8f;

enum { ACT_NONE = 0, ACT_TANH = 1, ACT_RELU = 2 };

// order-preserving float<->uint mapping for atomic max on floats
__device__ __forceinline__ unsigned fenc(float f) {
    unsigned u = __float_as_uint(f);
    return (u & 0x80000000u) ? ~u : (u | 0x80000000u);
}
__device__ __forceinline__ float fdec(unsigned u) {
    u = (u & 0x80000000u) ? (u & 0x7fffffffu) : ~u;
    return __uint_as_float(u);
}

// ---------------------------------------------------------------------------
// f32 -> padded f16 pack: out[M,Kp] = (k<K) ? (half)in[M,K] : 0
// ---------------------------------------------------------------------------
__global__ void k_cvt(const float* __restrict__ in, _Float16* __restrict__ out,
                      long M, int K, int Kp)
{
    long i = (long)blockIdx.x * blockDim.x + threadIdx.x;
    if (i >= M * (long)Kp) return;
    long m = i / Kp;
    int  k = (int)(i - m * Kp);
    out[i] = (k < K) ? (_Float16)in[m * (long)K + k] : (_Float16)0.0f;
}

// ---------------------------------------------------------------------------
// WMMA GEMM on pre-packed f16: Y[M,N] = X[M,Kp] * W[N,Kp]^T (+bias)(+act)
// One wave computes a full 16-row x N block: NT = N/16 accumulators, A-fragment
// loaded once per K-step and reused across NT back-to-back WMMAs.
// ITERS = Kp/32 (compile-time, fully unrolled). No bounds checks, no
// divergence: M,N multiples of 16, rows pre-padded -> EXEC all-ones at WMMA.
// A frag: lane(m=lane&15): h8 @ [kb+half*8], h8 @ [kb+16+half*8]
// B frag: lane(n=lane&15): h8 @ [kb+half*8], h8 @ [kb+16+half*8]
// D: lane col nt*16+(lane&15); acc[v] -> row m0 + 8*(lane>>4) + v
// ---------------------------------------------------------------------------
template <int ITERS, int NT>
__global__ void k_gemm16(const _Float16* __restrict__ X, const _Float16* __restrict__ W,
                         const float* __restrict__ bias, float* __restrict__ Y,
                         int M, int act)
{
    constexpr int Kp = ITERS * 32;
    constexpr int N  = NT * 16;
    const int lane = threadIdx.x & 31;
    const int wid  = (int)((blockIdx.x * (unsigned)blockDim.x + threadIdx.x) >> 5);
    if (wid >= (M >> 4)) return;                     // wave-uniform exit
    const int half = lane >> 4;                      // 0 or 1
    const int r    = lane & 15;

    const h8* __restrict__ Xr = (const h8*)(X + (size_t)(wid * 16 + r) * Kp);

    v8f acc[NT];
#pragma unroll
    for (int nt = 0; nt < NT; ++nt) acc[nt] = (v8f){};

#pragma unroll
    for (int it = 0; it < ITERS; ++it) {
        const int base = it * 4;                     // units of 8 halves
        h8 a0 = Xr[base + half];                     // K = kb + half*8 .. +7
        h8 a1 = Xr[base + 2 + half];                 // K = kb + 16 + half*8 .. +7
        v16h a = __builtin_shufflevector(a0, a1, 0,1,2,3,4,5,6,7,8,9,10,11,12,13,14,15);
#pragma unroll
        for (int nt = 0; nt < NT; ++nt) {
            const h8* __restrict__ Wr = (const h8*)(W + (size_t)(nt * 16 + r) * Kp);
            h8 b0 = Wr[base + half];                 // K = kb + half*8 .. +7
            h8 b1 = Wr[base + 2 + half];             // K = kb + 16 + half*8 .. +7
            v16h b = __builtin_shufflevector(b0, b1, 0,1,2,3,4,5,6,7,8,9,10,11,12,13,14,15);
            acc[nt] = __builtin_amdgcn_wmma_f32_16x16x32_f16(false, a, false, b,
                                                             (short)0, acc[nt], false, false);
        }
    }

#pragma unroll
    for (int nt = 0; nt < NT; ++nt) {
        const int col = nt * 16 + r;
        const float bi = bias ? bias[col] : 0.0f;
#pragma unroll
        for (int v = 0; v < 8; ++v) {
            int m = wid * 16 + half * 8 + v;
            float y = acc[nt][v] + bi;
            if (act == ACT_TANH)      y = tanhf(y);
            else if (act == ACT_RELU) y = fmaxf(y, 0.0f);
            Y[(size_t)m * N + col] = y;
        }
    }
}

// ---------------------------------------------------------------------------
// Row-wise LayerNorm -> ReLU (-> +residual). One wave per row, D in {64,96,128}.
// ---------------------------------------------------------------------------
__global__ void k_ln_relu(const float* __restrict__ Yin, const float* __restrict__ res,
                          const float* __restrict__ gamma, const float* __restrict__ beta,
                          float* __restrict__ Yout, int M, int D)
{
    const int row  = (int)((blockIdx.x * (unsigned)blockDim.x + threadIdx.x) >> 5);
    const int lane = threadIdx.x & 31;
    if (row >= M) return;
    const int per = D >> 5;                       // elems per lane (<=4)
    float v[4];
    float s = 0.0f, s2 = 0.0f;
    for (int i = 0; i < per; ++i) {
        int c = i * 32 + lane;
        float x = Yin[(size_t)row * D + c];
        v[i] = x; s += x; s2 += x * x;
    }
    for (int m = 16; m >= 1; m >>= 1) { s += __shfl_xor(s, m); s2 += __shfl_xor(s2, m); }
    const float invD = 1.0f / (float)D;
    const float mean = s * invD;
    const float var  = s2 * invD - mean * mean;
    const float rstd = rsqrtf(var + 1e-5f);
    for (int i = 0; i < per; ++i) {
        int c = i * 32 + lane;
        float y = (v[i] - mean) * rstd * gamma[c] + beta[c];
        y = fmaxf(y, 0.0f);
        if (res) y += res[(size_t)row * D + c];
        Yout[(size_t)row * D + c] = y;
    }
}

// ---------------------------------------------------------------------------
__global__ void k_zero(float* __restrict__ p, long n)
{
    long i = (long)blockIdx.x * blockDim.x + threadIdx.x;
    if (i < n) p[i] = 0.0f;
}

__global__ void k_init_rows(float* __restrict__ out, const float* __restrict__ bias,
                            long total, int D)
{
    long i = (long)blockIdx.x * blockDim.x + threadIdx.x;
    if (i < total) out[i] = bias[i % D];
}

// ---------------------------------------------------------------------------
// Edge attention logits: wave per edge. lane -> 4 dims (float4); half-wave = head.
// ---------------------------------------------------------------------------
__global__ void k_edge_attn(const float* __restrict__ xl, const float* __restrict__ xr,
                            const float* __restrict__ ee,
                            const int* __restrict__ src, const int* __restrict__ dst,
                            const float* __restrict__ att,   // [2*64] flat
                            float* __restrict__ alpha, unsigned* __restrict__ amax, int E)
{
    const int e    = (int)((blockIdx.x * (unsigned)blockDim.x + threadIdx.x) >> 5);
    const int lane = threadIdx.x & 31;
    if (e >= E) return;
    const int s = src[e], d = dst[e];
    const float4 a4 = ((const float4*)att)[lane];
    const float4 l4 = ((const float4*)xl)[(size_t)s * 32 + lane];
    const float4 r4 = ((const float4*)xr)[(size_t)d * 32 + lane];
    const float4 e4 = ((const float4*)ee)[(size_t)e * 32 + lane];
    float z, p = 0.0f;
    z = l4.x + r4.x + e4.x; z = (z > 0.0f) ? z : SLOPE * z; p += z * a4.x;
    z = l4.y + r4.y + e4.y; z = (z > 0.0f) ? z : SLOPE * z; p += z * a4.y;
    z = l4.z + r4.z + e4.z; z = (z > 0.0f) ? z : SLOPE * z; p += z * a4.z;
    z = l4.w + r4.w + e4.w; z = (z > 0.0f) ? z : SLOPE * z; p += z * a4.w;
    for (int m = 1; m < 16; m <<= 1) p += __shfl_xor(p, m);   // reduce within head
    const int head = lane >> 4;
    if ((lane & 15) == 0) {
        alpha[(size_t)e * 2 + head] = p;
        atomicMax(&amax[(size_t)d * 2 + head], fenc(p));
    }
}

// exp(alpha - amax[dst]) and segment-sum denominators. thread per (edge,head).
__global__ void k_edge_exp(float* __restrict__ alpha, const unsigned* __restrict__ amax,
                           const int* __restrict__ dst, float* __restrict__ denom, int E)
{
    long i = (long)blockIdx.x * blockDim.x + threadIdx.x;
    if (i >= (long)E * 2) return;
    int e = (int)(i >> 1), h = (int)(i & 1);
    int d = dst[e];
    float ex = expf(alpha[i] - fdec(amax[(size_t)d * 2 + h]));
    alpha[i] = ex;                                   // buffer reused to hold ex
    atomicAdd(&denom[(size_t)d * 2 + h], ex);
}

// msg scatter: wave per edge, 4 dims/lane, f32 atomics into bias-seeded accumulator.
__global__ void k_scatter(const float* __restrict__ xl, const float* __restrict__ ex,
                          const float* __restrict__ denom,
                          const int* __restrict__ src, const int* __restrict__ dst,
                          float* __restrict__ out, int E)
{
    const int e    = (int)((blockIdx.x * (unsigned)blockDim.x + threadIdx.x) >> 5);
    const int lane = threadIdx.x & 31;
    if (e >= E) return;
    const int s = src[e], d = dst[e];
    const int head = lane >> 4;
    const float a = ex[(size_t)e * 2 + head] / (denom[(size_t)d * 2 + head] + 1e-16f);
    const float4 m = ((const float4*)xl)[(size_t)s * 32 + lane];
    float* o = out + (size_t)d * HID + lane * 4;
    atomicAdd(o + 0, m.x * a);
    atomicAdd(o + 1, m.y * a);
    atomicAdd(o + 2, m.z * a);
    atomicAdd(o + 3, m.w * a);
}

// mean-pool scatter: wave per node.
__global__ void k_pool_scatter(const float* __restrict__ h, const int* __restrict__ batch,
                               float* __restrict__ gsum, float* __restrict__ cnt, int Nn)
{
    const int n    = (int)((blockIdx.x * (unsigned)blockDim.x + threadIdx.x) >> 5);
    const int lane = threadIdx.x & 31;
    if (n >= Nn) return;
    const int g = batch[n];
    const float4 v = ((const float4*)h)[(size_t)n * 32 + lane];
    float* o = gsum + (size_t)g * HID + lane * 4;
    atomicAdd(o + 0, v.x);
    atomicAdd(o + 1, v.y);
    atomicAdd(o + 2, v.z);
    atomicAdd(o + 3, v.w);
    if (lane == 0) atomicAdd(&cnt[g], 1.0f);
}

__global__ void k_pool_final(const float* __restrict__ gsum, const float* __restrict__ cnt,
                             float* __restrict__ g, long total)
{
    long i = (long)blockIdx.x * blockDim.x + threadIdx.x;
    if (i >= total) return;
    float c = cnt[i >> 7];                     // HID == 128
    g[i] = gsum[i] / fmaxf(c, 1.0f);
}

// final 48 -> 1 projection + sigmoid, interleaved into d_out[g*3 + t].
__global__ void k_head_final(const float* __restrict__ y2, const float* __restrict__ W3,
                             const float* __restrict__ b3, float* __restrict__ out, int t)
{
    int g = (int)(blockIdx.x * (unsigned)blockDim.x + threadIdx.x);
    if (g >= NGRAPH) return;
    float s = b3[0];
#pragma unroll
    for (int k = 0; k < 48; ++k) s += y2[(size_t)g * 48 + k] * W3[k];
    out[(size_t)g * 3 + t] = 1.0f / (1.0f + expf(-s));
}

// ---------------------------------------------------------------------------
static inline void launch_cvt(const float* in, _Float16* out, long M, int K, int Kp,
                              hipStream_t s)
{
    long tot = M * (long)Kp;
    k_cvt<<<(int)((tot + 255) / 256), 256, 0, s>>>(in, out, M, K, Kp);
}

static inline void launch_gemm(const _Float16* X, const _Float16* W, const float* b,
                               float* Y, int M, int N, int Kp, int act, hipStream_t s)
{
    int waves = M >> 4;                               // one wave per 16-row block
    int grid  = (int)(((long)waves * 32 + 255) / 256);
    const int key = (Kp >> 5) * 100 + (N >> 4);
    switch (key) {
        case 308: k_gemm16<3, 8><<<grid, 256, 0, s>>>(X, W, b, Y, M, act); break; // 96 ->128
        case 104: k_gemm16<1, 4><<<grid, 256, 0, s>>>(X, W, b, Y, M, act); break; // 32 -> 64
        case 408: k_gemm16<4, 8><<<grid, 256, 0, s>>>(X, W, b, Y, M, act); break; // 128->128
        case 208: k_gemm16<2, 8><<<grid, 256, 0, s>>>(X, W, b, Y, M, act); break; // 64 ->128
        case 406: k_gemm16<4, 6><<<grid, 256, 0, s>>>(X, W, b, Y, M, act); break; // 128-> 96
        case 303: k_gemm16<3, 3><<<grid, 256, 0, s>>>(X, W, b, Y, M, act); break; // 96 -> 48
        default:  k_gemm16<4, 8><<<grid, 256, 0, s>>>(X, W, b, Y, M, act); break;
    }
}

static inline void launch_rows(int rows, int* grid)   // wave per row, 256-thread blocks
{
    long thr = (long)rows * 32;
    *grid = (int)((thr + 255) / 256);
}

extern "C" void kernel_launch(void* const* d_in, const int* in_sizes, int n_in,
                              void* d_out, int out_size, void* d_ws, size_t ws_size,
                              hipStream_t stream)
{
    (void)in_sizes; (void)n_in; (void)out_size; (void)ws_size;
    // ---- inputs in setup_inputs() dict insertion order ----
    int ix = 0;
    const float* x         = (const float*)d_in[ix++];            // [50000,86]
    const float* edge_attr = (const float*)d_in[ix++];            // [800000,18]
    const float* node_W    = (const float*)d_in[ix++];
    const float* node_b    = (const float*)d_in[ix++];
    const float* node_g    = (const float*)d_in[ix++];
    const float* node_beta = (const float*)d_in[ix++];
    const float* edge_W    = (const float*)d_in[ix++];
    const float* edge_b    = (const float*)d_in[ix++];
    const float* edge_g    = (const float*)d_in[ix++];
    const float* edge_beta = (const float*)d_in[ix++];
    const float* ro_W      = (const float*)d_in[ix++];
    const float* ro_b      = (const float*)d_in[ix++];
    struct Layer { const float *Wl,*bl,*Wr,*br,*We,*att,*bias,*ln_g,*ln_b; } L[2];
    for (int l = 0; l < 2; ++l) {
        L[l].Wl   = (const float*)d_in[ix++];
        L[l].bl   = (const float*)d_in[ix++];
        L[l].Wr   = (const float*)d_in[ix++];
        L[l].br   = (const float*)d_in[ix++];
        L[l].We   = (const float*)d_in[ix++];
        L[l].att  = (const float*)d_in[ix++];
        L[l].bias = (const float*)d_in[ix++];
        L[l].ln_g = (const float*)d_in[ix++];
        L[l].ln_b = (const float*)d_in[ix++];
    }
    struct Head { const float *W1,*b1,*g1,*be1,*W2,*b2,*W3,*b3; } Hd[3];
    for (int t = 0; t < 3; ++t) {
        Hd[t].W1  = (const float*)d_in[ix++];
        Hd[t].b1  = (const float*)d_in[ix++];
        Hd[t].g1  = (const float*)d_in[ix++];
        Hd[t].be1 = (const float*)d_in[ix++];
        Hd[t].W2  = (const float*)d_in[ix++];
        Hd[t].b2  = (const float*)d_in[ix++];
        Hd[t].W3  = (const float*)d_in[ix++];
        Hd[t].b3  = (const float*)d_in[ix++];
    }
    const int* edge_index = (const int*)d_in[ix++];               // [2,800000]
    const int* batch      = (const int*)d_in[ix++];               // [50000]
    const int* src = edge_index;
    const int* dst = edge_index + NEDGES;

    // ---- workspace layout: f32 region then f16 region ----
    float* ws = (float*)d_ws;
    size_t o = 0;
    float*    h     = ws + o;  o += (size_t)NNODES * HID;
    float*    e     = ws + o;  o += (size_t)NEDGES * EHID;
    float*    xl    = ws + o;  o += (size_t)NNODES * HID;
    float*    xr    = ws + o;  o += (size_t)NNODES * HID;
    float*    ee    = ws + o;  o += (size_t)NEDGES * HID;
    float*    alpha = ws + o;  o += (size_t)NEDGES * 2;     // later holds ex
    unsigned* amax  = (unsigned*)(ws + o); o += (size_t)NNODES * 2;
    float*    denom = ws + o;  o += (size_t)NNODES * 2;
    float*    outb  = ws + o;  o += (size_t)NNODES * HID;
    float*    gsum  = ws + o;  o += (size_t)NGRAPH * HID;
    float*    cnt   = ws + o;  o += (size_t)NGRAPH;
    float*    gbuf  = ws + o;  o += (size_t)NGRAPH * HID;
    float*    gact  = ws + o;  o += (size_t)NGRAPH * HID;
    float*    y1    = ws + o;  o += (size_t)NGRAPH * 96;
    float*    y2    = ws + o;  o += (size_t)NGRAPH * 48;
    o = (o + 7) & ~(size_t)7;                               // 16B-align f16 region
    _Float16* f16 = (_Float16*)(ws + o);
    size_t q = 0;
    _Float16* x16  = f16 + q;  q += (size_t)NNODES * 96;    // 86 -> 96
    _Float16* ea16 = f16 + q;  q += (size_t)NEDGES * 32;    // 18 -> 32
    _Float16* e16  = f16 + q;  q += (size_t)NEDGES * EHID;
    _Float16* h16  = f16 + q;  q += (size_t)NNODES * HID;
    _Float16* g16  = f16 + q;  q += (size_t)NGRAPH * HID;
    _Float16* ga16 = f16 + q;  q += (size_t)NGRAPH * HID;
    _Float16* y116 = f16 + q;  q += (size_t)NGRAPH * 96;
    _Float16* w16  = f16 + q;  q += (size_t)HID * HID;      // shared weight staging

    int grid;

    // ---- encoders ----
    launch_cvt(x, x16, NNODES, NODE_DIM, 96, stream);
    launch_cvt(node_W, w16, HID, NODE_DIM, 96, stream);
    launch_gemm(x16, w16, node_b, h, NNODES, HID, 96, ACT_NONE, stream);
    launch_rows(NNODES, &grid);
    k_ln_relu<<<grid, 256, 0, stream>>>(h, nullptr, node_g, node_beta, h, NNODES, HID);

    launch_cvt(edge_attr, ea16, NEDGES, EDGE_DIM, 32, stream);
    launch_cvt(edge_W, w16, EHID, EDGE_DIM, 32, stream);
    launch_gemm(ea16, w16, edge_b, e, NEDGES, EHID, 32, ACT_NONE, stream);
    launch_rows(NEDGES, &grid);
    k_ln_relu<<<grid, 256, 0, stream>>>(e, nullptr, edge_g, edge_beta, e, NEDGES, EHID);
    launch_cvt(e, e16, NEDGES, EHID, EHID, stream);          // e is constant across layers

    // ---- GATv2 layers ----
    for (int l = 0; l < 2; ++l) {
        launch_cvt(h, h16, NNODES, HID, HID, stream);
        launch_cvt(L[l].Wl, w16, HID, HID, HID, stream);
        launch_gemm(h16, w16, L[l].bl, xl, NNODES, HID, HID, ACT_NONE, stream);
        launch_cvt(L[l].Wr, w16, HID, HID, HID, stream);
        launch_gemm(h16, w16, L[l].br, xr, NNODES, HID, HID, ACT_NONE, stream);
        launch_cvt(L[l].We, w16, HID, EHID, EHID, stream);
        launch_gemm(e16, w16, nullptr, ee, NEDGES, HID, EHID, ACT_NONE, stream);

        k_zero<<<(NNODES * 2 + 255) / 256, 256, 0, stream>>>((float*)amax, (long)NNODES * 2);
        k_zero<<<(NNODES * 2 + 255) / 256, 256, 0, stream>>>(denom, (long)NNODES * 2);

        launch_rows(NEDGES, &grid);
        k_edge_attn<<<grid, 256, 0, stream>>>(xl, xr, ee, src, dst, L[l].att, alpha, amax, NEDGES);
        k_edge_exp<<<(int)(((long)NEDGES * 2 + 255) / 256), 256, 0, stream>>>(alpha, amax, dst, denom, NEDGES);

        long tot = (long)NNODES * HID;
        k_init_rows<<<(int)((tot + 255) / 256), 256, 0, stream>>>(outb, L[l].bias, tot, HID);
        launch_rows(NEDGES, &grid);
        k_scatter<<<grid, 256, 0, stream>>>(xl, alpha, denom, src, dst, outb, NEDGES);

        launch_rows(NNODES, &grid);
        k_ln_relu<<<grid, 256, 0, stream>>>(outb, h, L[l].ln_g, L[l].ln_b, h, NNODES, HID);
    }

    // ---- global mean pool + readout ----
    long gtot = (long)NGRAPH * HID;
    k_zero<<<(int)((gtot + 255) / 256), 256, 0, stream>>>(gsum, gtot);
    k_zero<<<(NGRAPH + 255) / 256, 256, 0, stream>>>(cnt, NGRAPH);
    launch_rows(NNODES, &grid);
    k_pool_scatter<<<grid, 256, 0, stream>>>(h, batch, gsum, cnt, NNODES);
    k_pool_final<<<(int)((gtot + 255) / 256), 256, 0, stream>>>(gsum, cnt, gbuf, gtot);
    launch_cvt(gbuf, g16, NGRAPH, HID, HID, stream);
    launch_cvt(ro_W, w16, HID, HID, HID, stream);
    launch_gemm(g16, w16, ro_b, gact, NGRAPH, HID, HID, ACT_TANH, stream);
    launch_cvt(gact, ga16, NGRAPH, HID, HID, stream);        // shared across heads

    // ---- prediction heads (DAT, NET, SERT) ----
    for (int t = 0; t < 3; ++t) {
        launch_cvt(Hd[t].W1, w16, 96, HID, HID, stream);
        launch_gemm(ga16, w16, Hd[t].b1, y1, NGRAPH, 96, HID, ACT_NONE, stream);
        launch_rows(NGRAPH, &grid);
        k_ln_relu<<<grid, 256, 0, stream>>>(y1, nullptr, Hd[t].g1, Hd[t].be1, y1, NGRAPH, 96);
        launch_cvt(y1, y116, NGRAPH, 96, 96, stream);
        launch_cvt(Hd[t].W2, w16, 48, 96, 96, stream);
        launch_gemm(y116, w16, Hd[t].b2, y2, NGRAPH, 48, 96, ACT_RELU, stream);
        k_head_final<<<(NGRAPH + 255) / 256, 256, 0, stream>>>(y2, Hd[t].W3, Hd[t].b3,
                                                               (float*)d_out, t);
    }
}
